// GAT_10934986736301
// MI455X (gfx1250) — compile-verified, matrix operand order checked by reading, not run on hardware
//
#include <hip/hip_runtime.h>
#include <hip/hip_bf16.h>

// ---------------- problem constants (from reference) ----------------
#define GN       100000      // nodes
#define GE       1600000     // edges
#define IN_DIM   128
#define HID      64
#define NCLS     40
#define NCLS_PAD 48
#define NEG_BIG_F (-9000000000000000.0f)

typedef __bf16 bf16_t;
typedef bf16_t v16bf __attribute__((ext_vector_type(16)));
typedef float  v8f   __attribute__((ext_vector_type(8)));
typedef unsigned int u32x4 __attribute__((ext_vector_type(4)));
typedef int          i32x4 __attribute__((ext_vector_type(4)));
typedef int          i32x8 __attribute__((ext_vector_type(8)));

#if defined(__has_builtin)
#if __has_builtin(__builtin_amdgcn_tensor_load_to_lds) && \
    __has_builtin(__builtin_amdgcn_s_wait_tensorcnt)
#define GAT_HAVE_TDM 1
#endif
#endif
#ifndef GAT_HAVE_TDM
#define GAT_HAVE_TDM 0
#endif

// ---------------------------------------------------------------------------
// GEMM: C[M,ldc] = A[M,K] @ B[K,NB] (+bias), bf16 WMMA, f32 accumulate.
//   - one wave computes a 16 x (NT*16) output tile
//   - weights DMA'd global->LDS by the Tensor Data Mover (TENSORcnt), then
//     swizzled into exact WMMA B-fragment order as bf16 (one 32B ds read/frag)
//   - A fragment: per lane two contiguous 8-float runs -> 4x global_load_b128
// ---------------------------------------------------------------------------
template <int KSTEPS, int NT>
__global__ void gat_gemm_wmma_kernel(const float* __restrict__ A,
                                     const float* __restrict__ B,
                                     const float* __restrict__ bias,
                                     float* __restrict__ C,
                                     int M, int NB, int ldc) {
  constexpr int K    = KSTEPS * 32;
  constexpr int NT16 = NT * 16;
  // fragment store: [frag][lane][elem], frag = s*NT + t, 16 bf16 per lane
  __shared__ __align__(32) bf16_t ldsB[KSTEPS * NT * 32 * 16];

#if GAT_HAVE_TDM
  __shared__ __align__(16) float ldsW[K * 64];  // raw f32 weights (NB <= 64)
  if ((threadIdx.x >> 5) == 0) {                // one wave issues the TDM op
    const unsigned long long ga = (unsigned long long)(const void*)B;
    const unsigned ldsa = (unsigned)(uintptr_t)&ldsW[0];  // low 32b = LDS offset
    const unsigned T0 = (unsigned)(K * NB);     // contiguous f32 elements
    // D# group0: count=1 | lds_addr | global_addr[56:0] | type=2
    u32x4 g0;
    g0[0] = 1u;
    g0[1] = ldsa;
    g0[2] = (unsigned)(ga & 0xFFFFFFFFull);
    g0[3] = (unsigned)((ga >> 32) & 0x01FFFFFFull) | (2u << 30);
    // D# group1: data_size=4B, tensor_dim0=tile_dim0=T0, tensor_dim1=tile_dim1=1
    i32x8 g1;
    g1[0] = (int)(2u << 16);                          // data_size = 4 bytes
    g1[1] = (int)((T0 & 0xFFFFu) << 16);              // tensor_dim0[15:0]
    g1[2] = (int)(((T0 >> 16) & 0xFFFFu) | (1u << 16)); // dim0[31:16], tensor_dim1=1
    g1[3] = (int)((T0 & 0xFFFFu) << 16);              // tile_dim0 = T0 (<= 8192)
    g1[4] = 1;                                        // tile_dim1 = 1
    g1[5] = (int)T0;                                  // tensor_dim0_stride[31:0]
    g1[6] = 0;
    g1[7] = 0;
    const i32x4 gz = {0, 0, 0, 0};
#if __clang_major__ >= 23
    const i32x8 gz8 = {0, 0, 0, 0, 0, 0, 0, 0};
    __builtin_amdgcn_tensor_load_to_lds(g0, g1, gz, gz, gz8, 0);
#else
    __builtin_amdgcn_tensor_load_to_lds(g0, g1, gz, gz, 0);
#endif
    __builtin_amdgcn_s_wait_tensorcnt(0);
  }
  __syncthreads();
  const float* __restrict__ Bsrc = ldsW;
#else
  const float* __restrict__ Bsrc = B;
#endif

  // ---- block-cooperative swizzle of B into WMMA-fragment-order bf16 ----
  for (int idx = threadIdx.x; idx < K * NT16; idx += blockDim.x) {
    const int k = idx / NT16;
    const int n = idx - k * NT16;
    const int nsrc = (n < NB) ? n : (NB - 1);   // clamp padded tail columns
    const int s      = k >> 5;
    const int kk     = k & 31;
    const int hi     = kk >> 4;                 // element group (+16 K offset)
    const int rem    = kk & 15;
    const int halfk  = rem >> 3;                // lane half (0-15 vs 16-31)
    const int within = rem & 7;
    const int lane   = halfk * 16 + (n & 15);
    const int elem   = hi * 8 + within;
    const int frag   = s * NT + (n >> 4);
    ldsB[(frag * 32 + lane) * 16 + elem] = (bf16_t)Bsrc[k * NB + nsrc];
  }
  __syncthreads();

  const int tile = blockIdx.x * (blockDim.x >> 5) + (threadIdx.x >> 5);
  if (tile * 16 >= M) return;                   // wave-uniform (EXEC all-1)
  const int lane = threadIdx.x & 31;
  const int half = lane >> 4;
  const int col0 = lane & 15;

  const float* __restrict__ arow = A + ((long)tile * 16 + col0) * K;

  v8f acc[NT];
#pragma unroll
  for (int t = 0; t < NT; ++t) acc[t] = (v8f){0.f, 0.f, 0.f, 0.f, 0.f, 0.f, 0.f, 0.f};

#pragma unroll
  for (int s = 0; s < KSTEPS; ++s) {
    const int kbase = s * 32 + half * 8;
    if (s + 1 < KSTEPS)  // gfx1250 global_prefetch_b8 on next A k-slab
      __builtin_prefetch(arow + kbase + 32, 0, 0);
    // two contiguous 8-float runs per lane -> 4x global_load_b128
    const float4 f0 = *(const float4*)(arow + kbase);
    const float4 f1 = *(const float4*)(arow + kbase + 4);
    const float4 f2 = *(const float4*)(arow + kbase + 16);
    const float4 f3 = *(const float4*)(arow + kbase + 20);
    v16bf a;
    a[0]  = (bf16_t)f0.x; a[1]  = (bf16_t)f0.y; a[2]  = (bf16_t)f0.z; a[3]  = (bf16_t)f0.w;
    a[4]  = (bf16_t)f1.x; a[5]  = (bf16_t)f1.y; a[6]  = (bf16_t)f1.z; a[7]  = (bf16_t)f1.w;
    a[8]  = (bf16_t)f2.x; a[9]  = (bf16_t)f2.y; a[10] = (bf16_t)f2.z; a[11] = (bf16_t)f2.w;
    a[12] = (bf16_t)f3.x; a[13] = (bf16_t)f3.y; a[14] = (bf16_t)f3.z; a[15] = (bf16_t)f3.w;
    // hoist all NT B-fragments first so ds_loads issue as one clause
    v16bf bfr[NT];
#pragma unroll
    for (int t = 0; t < NT; ++t)
      bfr[t] = *(const v16bf*)&ldsB[((s * NT + t) * 32 + lane) * 16];
#pragma unroll
    for (int t = 0; t < NT; ++t)
      acc[t] = __builtin_amdgcn_wmma_f32_16x16x32_bf16(
          false, a, false, bfr[t], (short)0, acc[t], false, false);
  }

  // C/D layout: VGPR r holds M = r + 8*half; column = t*16 + col0
#pragma unroll
  for (int t = 0; t < NT; ++t) {
    const int n = t * 16 + col0;
    const int nb = (n < NB) ? n : (NB - 1);
    const float bv = bias ? bias[nb] : 0.f;
#pragma unroll
    for (int r = 0; r < 8; ++r) {
      const long m = (long)tile * 16 + half * 8 + r;
      C[m * (long)ldc + n] = acc[t][r] + bv;
    }
  }
}

// ---------------- per-node attention scores ----------------
__global__ void gat_sdots_kernel(const float* __restrict__ wh,
                                 const float* __restrict__ a_src,
                                 const float* __restrict__ a_dst,
                                 float* __restrict__ ssrc, float* __restrict__ sdst,
                                 int n) {
  const int i = blockIdx.x * blockDim.x + threadIdx.x;
  if (i >= n) return;
  const float4* row = (const float4*)(wh + (long)i * HID);
  const float4* as = (const float4*)a_src;
  const float4* ad = (const float4*)a_dst;
  float ss = 0.f, sd = 0.f;
#pragma unroll
  for (int j = 0; j < HID / 4; ++j) {
    const float4 v = row[j];
    const float4 s = as[j];
    const float4 d = ad[j];
    ss += v.x * s.x + v.y * s.y + v.z * s.z + v.w * s.w;
    sd += v.x * d.x + v.y * d.y + v.z * d.z + v.w * d.w;
  }
  ssrc[i] = ss;
  sdst[i] = sd;
}

// ---------------- edge softmax (global, axis=0) in 3 passes ----------------
__device__ __forceinline__ float edge_logit(const long long* __restrict__ ei,
                                            const float* __restrict__ ssrc,
                                            const float* __restrict__ sdst, int j) {
  const int s = (int)ei[j];
  const int d = (int)ei[GE + j];
  const float e = ssrc[s] + sdst[d];
  // leaky_relu then where(>0, e, NEG_BIG) collapses to this:
  return (e > 0.f) ? e : NEG_BIG_F;
}

__device__ __forceinline__ void atomic_max_f32(float* addr, float val) {
  unsigned int* ua = (unsigned int*)addr;
  unsigned int old = __float_as_uint(*addr);
  while (__uint_as_float(old) < val) {
    const unsigned int assumed = old;
    old = atomicCAS(ua, assumed, __float_as_uint(val));
    if (old == assumed) break;
  }
}

__global__ void gat_init_scal_kernel(float* scal) {
  if (threadIdx.x == 0 && blockIdx.x == 0) {
    scal[0] = NEG_BIG_F;  // running max
    scal[1] = 0.f;        // running sum(exp)
  }
}

__global__ void gat_edge_max_kernel(const long long* __restrict__ ei,
                                    const float* __restrict__ ssrc,
                                    const float* __restrict__ sdst,
                                    float* __restrict__ scal) {
  const int j = blockIdx.x * blockDim.x + threadIdx.x;
  float v = (j < GE) ? edge_logit(ei, ssrc, sdst, j) : NEG_BIG_F;
#pragma unroll
  for (int off = 16; off > 0; off >>= 1) v = fmaxf(v, __shfl_xor(v, off));
  if ((threadIdx.x & 31) == 0) atomic_max_f32(scal, v);
}

__global__ void gat_edge_sum_kernel(const long long* __restrict__ ei,
                                    const float* __restrict__ ssrc,
                                    const float* __restrict__ sdst,
                                    float* __restrict__ scal) {
  const int j = blockIdx.x * blockDim.x + threadIdx.x;
  const float gmax = scal[0];
  float v = 0.f;
  if (j < GE) v = __expf(edge_logit(ei, ssrc, sdst, j) - gmax);
#pragma unroll
  for (int off = 16; off > 0; off >>= 1) v += __shfl_xor(v, off);
  if ((threadIdx.x & 31) == 0) atomicAdd(scal + 1, v);
}

__global__ void gat_edge_scatter_kernel(const long long* __restrict__ ei,
                                        const float* __restrict__ ssrc,
                                        const float* __restrict__ sdst,
                                        const float* __restrict__ wh,
                                        const float* __restrict__ scal,
                                        float* __restrict__ hp) {
  const int j = blockIdx.x * blockDim.x + threadIdx.x;
  if (j >= GE) return;
  const int s = (int)ei[j];
  const int d = (int)ei[GE + j];
  const float e = ssrc[s] + sdst[d];
  const float logit = (e > 0.f) ? e : NEG_BIG_F;
  const float att = __expf(logit - scal[0]) / scal[1];
  const float4* wr = (const float4*)(wh + (long)s * HID);
  float* orow = hp + (long)d * HID;
#pragma unroll
  for (int q = 0; q < HID / 4; ++q) {
    const float4 v = wr[q];
    unsafeAtomicAdd(orow + 4 * q + 0, att * v.x);  // global_atomic_add_f32
    unsafeAtomicAdd(orow + 4 * q + 1, att * v.y);
    unsafeAtomicAdd(orow + 4 * q + 2, att * v.z);
    unsafeAtomicAdd(orow + 4 * q + 3, att * v.w);
  }
}

// ---------------- elementwise kernels ----------------
__global__ void gat_fill0_kernel(float* __restrict__ p, long n) {
  const long i = (long)blockIdx.x * blockDim.x + threadIdx.x;
  if (i < n) p[i] = 0.f;
}

__global__ void gat_elu_kernel(float* __restrict__ p, long n) {
  const long i = (long)blockIdx.x * blockDim.x + threadIdx.x;
  if (i < n) {
    const float x = p[i];
    p[i] = (x > 0.f) ? x : (__expf(x) - 1.f);
  }
}

// ---------------- final ELU + log_softmax over classes ----------------
__global__ void gat_out_lsm_kernel(const float* __restrict__ z, float* __restrict__ out, int n) {
  const int i = blockIdx.x * blockDim.x + threadIdx.x;
  if (i >= n) return;
  float v[NCLS];
  float mx = -3.4e38f;
#pragma unroll
  for (int j = 0; j < NCLS; ++j) {
    float t = z[(long)i * NCLS_PAD + j];
    t = (t > 0.f) ? t : (__expf(t) - 1.f);  // elu
    v[j] = t;
    mx = fmaxf(mx, t);
  }
  float sum = 0.f;
#pragma unroll
  for (int j = 0; j < NCLS; ++j) sum += __expf(v[j] - mx);
  const float lse = mx + __logf(sum);
#pragma unroll
  for (int j = 0; j < NCLS; ++j) out[(long)i * NCLS + j] = v[j] - lse;
}

// ---------------- driver ----------------
extern "C" void kernel_launch(void* const* d_in, const int* in_sizes, int n_in,
                              void* d_out, int out_size, void* d_ws, size_t ws_size,
                              hipStream_t stream) {
  (void)in_sizes; (void)n_in; (void)out_size; (void)ws_size;
  const float*     x     = (const float*)d_in[0];
  const long long* ei    = (const long long*)d_in[1];   // int64 [2, E]
  const float*     emb_w = (const float*)d_in[2];
  const float*     emb_b = (const float*)d_in[3];
  const float*     Ws    = (const float*)d_in[4];       // [2, 64, 64]
  const float*     attn  = (const float*)d_in[5];       // [2, 1, 128]
  const float*     out_w = (const float*)d_in[6];       // [64, 40]
  const float*     out_b = (const float*)d_in[7];
  float*           out   = (float*)d_out;

  // workspace layout
  float* h    = (float*)d_ws;            // [N,64]  (also scatter accumulator)
  float* wh   = h + (long)GN * HID;      // [N,64]
  float* ssrc = wh + (long)GN * HID;     // [N]
  float* sdst = ssrc + GN;               // [N]
  float* z    = sdst + GN;               // [N,48]
  float* scal = z + (long)GN * NCLS_PAD; // [2]

  const int TB = 256;
  const dim3 blk(TB);
  const int gemm_blocks = (GN / 16 + (TB / 32) - 1) / (TB / 32);  // 1 wave = 1 row-tile
  const int node_blocks = (GN + TB - 1) / TB;
  const int elem_blocks = ((long)GN * HID + TB - 1) / TB;
  const int edge_blocks = (GE + TB - 1) / TB;

  // 1) h = x @ emb_w + emb_b        (K=128 -> 4 k-steps, 4 n-tiles)
  gat_gemm_wmma_kernel<4, 4><<<gemm_blocks, blk, 0, stream>>>(
      x, emb_w, emb_b, h, GN, HID, HID);

  // 2) two GAT layers
  for (int l = 0; l < 2; ++l) {
    const float* W     = Ws + (long)l * HID * HID;
    const float* a_src = attn + (long)l * 2 * HID;
    const float* a_dst = a_src + HID;

    gat_gemm_wmma_kernel<2, 4><<<gemm_blocks, blk, 0, stream>>>(
        h, W, nullptr, wh, GN, HID, HID);

    gat_sdots_kernel<<<node_blocks, blk, 0, stream>>>(wh, a_src, a_dst, ssrc, sdst, GN);

    gat_init_scal_kernel<<<1, 32, 0, stream>>>(scal);
    gat_edge_max_kernel<<<edge_blocks, blk, 0, stream>>>(ei, ssrc, sdst, scal);
    gat_edge_sum_kernel<<<edge_blocks, blk, 0, stream>>>(ei, ssrc, sdst, scal);

    gat_fill0_kernel<<<elem_blocks, blk, 0, stream>>>(h, (long)GN * HID);
    gat_edge_scatter_kernel<<<edge_blocks, blk, 0, stream>>>(ei, ssrc, sdst, wh, scal, h);
    gat_elu_kernel<<<elem_blocks, blk, 0, stream>>>(h, (long)GN * HID);
  }

  // 3) z = h @ out_w + out_b (padded to 48 cols), then ELU + log_softmax
  gat_gemm_wmma_kernel<2, 3><<<gemm_blocks, blk, 0, stream>>>(
      h, out_w, out_b, z, GN, NCLS, NCLS_PAD);
  gat_out_lsm_kernel<<<node_blocks, blk, 0, stream>>>(z, out, GN);
}
